// BahdanauAttention_54013508715311
// MI455X (gfx1250) — matrix operand order repaired and checked by reading, not compiled, not run on hardware
//
#include <hip/hip_runtime.h>
#include <cstdint>
#include <cstddef>

#define EMBED 512
#define QDIM  512
#define VDIM  512
#define SRC   4096
#define BSZ   64
#define MROWS (SRC * BSZ) /* 262144 */

typedef __bf16 bf16_t;
typedef bf16_t v16bf __attribute__((ext_vector_type(16)));
typedef float  v8f   __attribute__((ext_vector_type(8)));

/* workspace layout (bytes) */
#define WS_NV    0                       /* 512 f32            */
#define WS_PQBT  2048                    /* 512x64 f32         */
#define WS_BT    (2048 + 131072)         /* 512x512 bf16 (WvT) */
#define WS_PART  (WS_BT + 524288)        /* 8x64x512 f32       */
#define NPART    8

/* LDS layout for k_logits */
#define A_STRIDE 1040                    /* 512 bf16 + 16B pad */
#define A_BYTES  (128 * A_STRIDE)        /* 133120 */
#define B_STRIDE 136                     /* 64 bf16 + 8B pad   */
#define B_BYTES  (512 * B_STRIDE)        /* 69632  */
#define SMEM_LOGITS (A_BYTES + B_BYTES)  /* 202752 */

__device__ __forceinline__ unsigned short f2bf(float f) {
  unsigned u = __builtin_bit_cast(unsigned, f);
  u += 0x7FFFu + ((u >> 16) & 1u);      /* round-to-nearest-even */
  return (unsigned short)(u >> 16);
}

/* hardware V_TANH_F32 when available (gfx1250 TRANS op), else libm */
__device__ __forceinline__ float htanh(float x) {
#if __has_builtin(__builtin_amdgcn_tanhf)
  return __builtin_amdgcn_tanhf(x);
#else
  return tanhf(x);
#endif
}

/* async global->LDS 16B copy; instruction offset applies to BOTH addresses */
#define ASYNC_CP16(la, ga, off)                                              \
  asm volatile("global_load_async_to_lds_b128 %0, %1, off offset:" #off     \
               :: "v"(la), "v"(ga) : "memory")

/* ---- prep: normed_v = g * v / ||v|| ------------------------------------ */
__global__ void k_prep_nv(const float* __restrict__ v,
                          const float* __restrict__ g,
                          float* __restrict__ nv) {
  __shared__ float red[256];
  const int t = threadIdx.x;
  float a = v[t], c = v[t + 256];
  red[t] = a * a + c * c;
  __syncthreads();
  for (int o = 128; o > 0; o >>= 1) {
    if (t < o) red[t] += red[t + o];
    __syncthreads();
  }
  const float inv = g[0] / sqrtf(red[0]);
  nv[t] = a * inv;
  nv[t + 256] = c * inv;
}

/* ---- prep: pqbT[e][b] = (query @ Wq^T)[b][e] + bias[e] ----------------- */
__global__ void k_prep_pqbT(const float* __restrict__ query,
                            const float* __restrict__ Wq,
                            const float* __restrict__ bias,
                            float* __restrict__ pqbT) {
  const int gid = blockIdx.x * 256 + threadIdx.x; /* 32768 total */
  const int e = gid >> 6, b = gid & 63;
  const float* q = query + b * QDIM;
  const float* w = Wq + e * QDIM;
  float acc = bias[e];
#pragma unroll 8
  for (int k = 0; k < QDIM; ++k) acc = fmaf(q[k], w[k], acc);
  pqbT[e * BSZ + b] = acc;
}

/* ---- prep: Bt[k][n] = bf16(Wv[n][k])  (row-major K x N) ---------------- */
__global__ void k_prep_bt(const float* __restrict__ Wv,
                          unsigned short* __restrict__ Bt) {
  const int idx = blockIdx.x * 256 + threadIdx.x; /* 262144 total */
  const int k = idx >> 9, n = idx & 511;
  Bt[idx] = f2bf(Wv[n * VDIM + k]);
}

/* ---- main WMMA kernel: logits[m] = nv . tanh(pqb + value[m]@Wv^T) ------ */
__global__ void __launch_bounds__(256, 2)
k_logits(const float* __restrict__ value,
         const unsigned short* __restrict__ Bt,
         const float* __restrict__ pqbT,
         const float* __restrict__ nv,
         float* __restrict__ logits) {
  extern __shared__ char smem[];
  char* Ab = smem;
  char* Bb = smem + A_BYTES;
  const int t = threadIdx.x;
  const int lane = t & 31, wave = t >> 5;
  const int m0 = blockIdx.x * 128;

  /* Stage A: 128 rows of value, fp32 -> bf16 into LDS */
  {
    const int r = t >> 1, h = t & 1;
    const float4* src = (const float4*)(value + (size_t)(m0 + r) * VDIM + h * 256);
    char* dst = Ab + r * A_STRIDE + h * 512;
#pragma unroll 4
    for (int c = 0; c < 64; ++c) {
      float4 f = src[c];
      uint2 p;
      p.x = (unsigned)f2bf(f.x) | ((unsigned)f2bf(f.y) << 16);
      p.y = (unsigned)f2bf(f.z) | ((unsigned)f2bf(f.w) << 16);
      *(uint2*)(dst + c * 8) = p;
    }
  }

  float racc[8] = {0, 0, 0, 0, 0, 0, 0, 0};
  const int half = lane >> 4;
  const int arow = (wave * 16 + (lane & 15)) * A_STRIDE;
  const int bb = (wave & 3) * 16 + half * 8; /* b-column base in pqbT */
  const unsigned lds_b = (unsigned)(size_t)(void*)Bb;

  for (int nc = 0; nc < 8; ++nc) {
    __syncthreads();
    /* stage B chunk (512 K-rows x 64 bf16) via async global->LDS DMA */
    for (int pass = 0; pass < 2; ++pass) {
      const int row = pass * 256 + t;
      const unsigned long long ga =
          (unsigned long long)(const void*)(Bt + (size_t)row * EMBED + nc * 64);
      const unsigned la = lds_b + row * B_STRIDE;
      ASYNC_CP16(la, ga, 0);
      ASYNC_CP16(la, ga, 16);
      ASYNC_CP16(la, ga, 32);
      ASYNC_CP16(la, ga, 48);
      ASYNC_CP16(la, ga, 64);
      ASYNC_CP16(la, ga, 80);
      ASYNC_CP16(la, ga, 96);
      ASYNC_CP16(la, ga, 112);
    }
    asm volatile("s_wait_asynccnt 0" ::: "memory");
    __syncthreads();

    for (int nt = 0; nt < 4; ++nt) {
      v8f acc = {0, 0, 0, 0, 0, 0, 0, 0};
#pragma unroll
      for (int kk = 0; kk < 16; ++kk) {
        union { v16bf v; uint4 u[2]; } af, bfm;
        const char* ap = Ab + arow + (kk * 32 + half * 8) * 2;
        af.u[0] = *(const uint4*)(ap);
        af.u[1] = *(const uint4*)(ap + 32);
        const char* bp = Bb + (nt * 16 + (lane & 15)) * 2;
#pragma unroll
        for (int s = 0; s < 16; ++s) {
          const int kr = kk * 32 + (s < 8 ? half * 8 + s : 16 + half * 8 + (s - 8));
          bfm.v[s] = *(const bf16_t*)(bp + kr * B_STRIDE);
        }
        acc = __builtin_amdgcn_wmma_f32_16x16x32_bf16(
            false, af.v, false, bfm.v, (short)0, acc, false, false);
      }
      const int e = nc * 64 + nt * 16 + (lane & 15);
      const float nve = nv[e];
      const float* pp = pqbT + e * BSZ + bb;
      float4 p0 = *(const float4*)pp;
      float4 p1 = *(const float4*)(pp + 4);
      racc[0] += htanh(acc[0] + p0.x) * nve;
      racc[1] += htanh(acc[1] + p0.y) * nve;
      racc[2] += htanh(acc[2] + p0.z) * nve;
      racc[3] += htanh(acc[3] + p0.w) * nve;
      racc[4] += htanh(acc[4] + p1.x) * nve;
      racc[5] += htanh(acc[5] + p1.y) * nve;
      racc[6] += htanh(acc[6] + p1.z) * nve;
      racc[7] += htanh(acc[7] + p1.w) * nve;
    }
  }

  /* reduce over the 16 N-lanes of each half-wave; C layout: M = j + 8*half */
#pragma unroll
  for (int j = 0; j < 8; ++j) {
    float s = racc[j];
    s += __shfl_xor(s, 8, 16);
    s += __shfl_xor(s, 4, 16);
    s += __shfl_xor(s, 2, 16);
    s += __shfl_xor(s, 1, 16);
    if ((lane & 15) == 0)
      logits[m0 + wave * 16 + half * 8 + j] = s; /* flat m == s*64+b */
  }
}

/* ---- softmax over s (axis 0) per batch column b ------------------------ */
__global__ void k_softmax(const unsigned char* __restrict__ mask,
                          float* __restrict__ attn1,
                          float* __restrict__ attn2) {
  __shared__ float red[256];
  const int b = blockIdx.x, t = threadIdx.x;
  float x[16];
  float mx = -INFINITY;
#pragma unroll
  for (int i = 0; i < 16; ++i) {
    const int s = i * 256 + t;
    float v = attn1[s * BSZ + b];
    if (mask[s * BSZ + b]) v = -INFINITY;
    x[i] = v;
    mx = fmaxf(mx, v);
  }
  red[t] = mx;
  __syncthreads();
  for (int o = 128; o > 0; o >>= 1) {
    if (t < o) red[t] = fmaxf(red[t], red[t + o]);
    __syncthreads();
  }
  mx = red[0];
  __syncthreads();
  float sum = 0.f;
#pragma unroll
  for (int i = 0; i < 16; ++i) {
    x[i] = __expf(x[i] - mx);
    sum += x[i];
  }
  red[t] = sum;
  __syncthreads();
  for (int o = 128; o > 0; o >>= 1) {
    if (t < o) red[t] += red[t + o];
    __syncthreads();
  }
  const float inv = 1.f / red[0];
#pragma unroll
  for (int i = 0; i < 16; ++i) {
    const int idx = (i * 256 + t) * BSZ + b;
    const float y = x[i] * inv;
    attn1[idx] = y;
    attn2[idx] = y;
  }
}

/* ---- context partials: part[sc][b][vd] = sum_{s in chunk} attn*value --- */
__global__ void k_ctx_part(const float* __restrict__ value,
                           const float* __restrict__ attn,
                           float* __restrict__ part) {
  const int sc = blockIdx.x >> 6, b = blockIdx.x & 63;
  const int t = threadIdx.x;
  float a0 = 0.f, a1 = 0.f;
  const int s0 = sc * (SRC / NPART);
  for (int s = s0; s < s0 + SRC / NPART; ++s) {
    const int m = s * BSZ + b;
    const float a = attn[m];
    const float* vr = value + (size_t)m * VDIM;
    /* pull next iteration's row toward the caches while the FMAs run */
    __builtin_prefetch(vr + BSZ * VDIM, 0, 1);
    a0 = fmaf(a, vr[t], a0);
    a1 = fmaf(a, vr[t + 256], a1);
  }
  float* pr = part + ((size_t)sc * BSZ + b) * VDIM;
  pr[t] = a0;
  pr[t + 256] = a1;
}

__global__ void k_ctx_reduce(const float* __restrict__ part,
                             float* __restrict__ ctx) {
  const int gid = blockIdx.x * 256 + threadIdx.x; /* 32768 = b*512+vd */
  float s = 0.f;
#pragma unroll
  for (int sc = 0; sc < NPART; ++sc) s += part[sc * (BSZ * VDIM) + gid];
  ctx[gid] = s;
}

extern "C" void kernel_launch(void* const* d_in, const int* in_sizes, int n_in,
                              void* d_out, int out_size, void* d_ws, size_t ws_size,
                              hipStream_t stream) {
  const float* query = (const float*)d_in[0];
  const float* value = (const float*)d_in[1];
  const unsigned char* mask = (const unsigned char*)d_in[2];
  const float* Wq = (const float*)d_in[3];
  const float* Wv = (const float*)d_in[4];
  const float* v = (const float*)d_in[5];
  const float* bias = (const float*)d_in[6];
  const float* g = (const float*)d_in[7];

  char* ws = (char*)d_ws;
  float* nv = (float*)(ws + WS_NV);
  float* pqbT = (float*)(ws + WS_PQBT);
  unsigned short* Bt = (unsigned short*)(ws + WS_BT);
  float* part = (float*)(ws + WS_PART);

  float* out = (float*)d_out;
  float* ctx = out;                        /* [64, 512]  */
  float* attn1 = out + BSZ * VDIM;         /* [4096, 64] */
  float* attn2 = attn1 + SRC * BSZ;        /* [4096, 64] */

  k_prep_nv<<<1, 256, 0, stream>>>(v, g, nv);
  k_prep_pqbT<<<(BSZ * EMBED) / 256, 256, 0, stream>>>(query, Wq, bias, pqbT);
  k_prep_bt<<<(EMBED * VDIM) / 256, 256, 0, stream>>>(Wv, Bt);
  k_logits<<<MROWS / 128, 256, SMEM_LOGITS, stream>>>(value, Bt, pqbT, nv, attn1);
  k_softmax<<<BSZ, 256, 0, stream>>>(mask, attn1, attn2);
  k_ctx_part<<<NPART * BSZ, 256, 0, stream>>>(value, attn1, part);
  k_ctx_reduce<<<(BSZ * VDIM) / 256, 256, 0, stream>>>(part, ctx);
}
